// GCNEncoder_81200651698600
// MI455X (gfx1250) — matrix operand order repaired and checked by reading, not compiled, not run on hardware
//
#include <hip/hip_runtime.h>
#include <hip/hip_bf16.h>

#define FEAT 128

typedef __attribute__((ext_vector_type(16))) __bf16 v16bf;
typedef __attribute__((ext_vector_type(8)))  float  v8f;

union BFrag { unsigned int u[8]; v16bf v; };

__device__ __forceinline__ unsigned int f2bf(float f) {
  unsigned int u = __float_as_uint(f);
  unsigned int r = u + 0x7FFFu + ((u >> 16) & 1u);  // RNE
  return r >> 16;
}

__device__ __forceinline__ unsigned int pk2bf(float a, float b) {
#if __has_builtin(__builtin_amdgcn_cvt_pk_bf16_f32)
  auto t = __builtin_amdgcn_cvt_pk_bf16_f32(a, b);   // v_cvt_pk_bf16_f32
  union { decltype(t) x; unsigned int u; } cv;
  cv.x = t;
  return cv.u;
#else
  return f2bf(a) | (f2bf(b) << 16);
#endif
}

// ---------------------------------------------------------------------------
// GEMM: H[nrows x 128] = (RELU ? relu(X) : X)[nrows x 128] @ W[128 x 128]
// bf16 WMMA, f32 accumulate. Block = 256 threads = 8 waves = 128 rows.
//  - Wt at LDS offset 0 (bf16, transposed) => all B loads are one base VGPR
//    + immediate DS offsets (ds_load_b128 pairs).
//  - B fragments ping-ponged so WMMA overlaps the next fragment's loads.
//  - D staged per-wave in LDS, written back as coalesced b128 with a
//    wave-uniform tail check.
// ---------------------------------------------------------------------------
template <int RELU>
__global__ __launch_bounds__(256) void gemm_wmma_bf16(
    const float* __restrict__ X, const float* __restrict__ W,
    float* __restrict__ H, int nrows)
{
  __shared__ __align__(16) char smem[(32 + 64) * 1024];
  unsigned short* Wt  = (unsigned short*)smem;          // 32 KB bf16 Wt[n][k]
  float*          Dst = (float*)(smem + 32 * 1024);     // 64 KB, 8 waves x 16x128
  const int tid = threadIdx.x;

  // Stage + transpose + convert W: Wt[n*128 + k] = bf16(W[k*128 + n])
  for (int i = tid; i < FEAT * FEAT; i += 256) {
    int k = i >> 7, n = i & 127;
    Wt[n * FEAT + k] = (unsigned short)f2bf(W[i]);
  }
  __syncthreads();

  const int wave  = tid >> 5;
  const int lane  = tid & 31;
  const int l16   = lane & 15;
  const int lhalf = lane >> 4;                 // 0 or 1

  const int tile_row = blockIdx.x * 128 + wave * 16;
  int row  = tile_row + l16;
  int rowc = row < nrows ? row : (nrows - 1);  // clamp: EXEC must stay full for WMMA
  const float* xr = X + (size_t)rowc * FEAT;

  const int kbase = lhalf * 16;                // A-fragment K base per lane half
  // Single base pointer for ALL B loads: dword index l16*64 + lhalf*8;
  // element (nt, kc, v) at immediate offset nt*1024 + kc*16 + v dwords.
  const unsigned int* bptr = (const unsigned int*)Wt + l16 * (FEAT / 2) + lhalf * 8;

  v8f acc[8] = {};                             // 8 N-tiles of 16 cols

#pragma unroll
  for (int kc = 0; kc < 4; ++kc) {             // K = 128 in chunks of 32
    const int kk0 = kc * 32;

    // A fragment (16x32 bf16): lane K set = {kb..kb+15}
    const float4* xa = (const float4*)(xr + kk0 + kbase);
    float4 p0 = xa[0];
    float4 p1 = xa[1];
    float4 p2 = xa[2];                          // +8 floats
    float4 p3 = xa[3];
    if (RELU) {
      p0.x = fmaxf(p0.x, 0.f); p0.y = fmaxf(p0.y, 0.f); p0.z = fmaxf(p0.z, 0.f); p0.w = fmaxf(p0.w, 0.f);
      p1.x = fmaxf(p1.x, 0.f); p1.y = fmaxf(p1.y, 0.f); p1.z = fmaxf(p1.z, 0.f); p1.w = fmaxf(p1.w, 0.f);
      p2.x = fmaxf(p2.x, 0.f); p2.y = fmaxf(p2.y, 0.f); p2.z = fmaxf(p2.z, 0.f); p2.w = fmaxf(p2.w, 0.f);
      p3.x = fmaxf(p3.x, 0.f); p3.y = fmaxf(p3.y, 0.f); p3.z = fmaxf(p3.z, 0.f); p3.w = fmaxf(p3.w, 0.f);
    }
    BFrag A;
    A.u[0] = pk2bf(p0.x, p0.y); A.u[1] = pk2bf(p0.z, p0.w);
    A.u[2] = pk2bf(p1.x, p1.y); A.u[3] = pk2bf(p1.z, p1.w);
    A.u[4] = pk2bf(p2.x, p2.y); A.u[5] = pk2bf(p2.z, p2.w);
    A.u[6] = pk2bf(p3.x, p3.y); A.u[7] = pk2bf(p3.z, p3.w);

    // Ping-pong B fragments over the 8 N-tiles
    BFrag Bb[2];
#pragma unroll
    for (int v = 0; v < 8; ++v) Bb[0].u[v] = bptr[kc * 16 + v];
#pragma unroll
    for (int nt = 0; nt < 8; ++nt) {
      if (nt < 7) {
#pragma unroll
        for (int v = 0; v < 8; ++v)
          Bb[(nt + 1) & 1].u[v] = bptr[(nt + 1) * 1024 + kc * 16 + v];
      }
      acc[nt] = __builtin_amdgcn_wmma_f32_16x16x32_bf16(
          false, A.v, false, Bb[nt & 1].v, (short)0, acc[nt], false, false);
    }
  }

  // Stage D in this wave's LDS region (same-wave DS ordering; no barrier)
  float* dw = Dst + wave * 16 * FEAT;
#pragma unroll
  for (int nt = 0; nt < 8; ++nt) {
    const int col = nt * 16 + l16;
#pragma unroll
    for (int r = 0; r < 8; ++r) {
      dw[(r + lhalf * 8) * FEAT + col] = acc[nt][r];
    }
  }

  // Coalesced b128 writeback; row bound is wave-uniform -> scalar branch
#pragma unroll
  for (int i = 0; i < 16; ++i) {
    int orow = tile_row + i;
    if (orow < nrows) {
      float4 v = ((const float4*)(dw + i * FEAT))[lane];
      *(float4*)(H + (size_t)orow * FEAT + lane * 4) = v;
    }
  }
}

// ---------------------------------------------------------------------------
// Degree / norm kernels
// ---------------------------------------------------------------------------
__global__ void k_fill_ones(float* __restrict__ p, int n) {
  int i = blockIdx.x * blockDim.x + threadIdx.x;
  if (i < n) p[i] = 1.0f;                       // self-loop contribution
}

__global__ void k_deg_acc(const int* __restrict__ ei, float* __restrict__ deg, int E) {
  int i = blockIdx.x * blockDim.x + threadIdx.x;
  if (i < E) atomicAdd(&deg[ei[E + i]], 1.0f);  // dst = ei[1][e]
}

__global__ void k_rsqrt(float* __restrict__ p, int n) {
  int i = blockIdx.x * blockDim.x + threadIdx.x;
  if (i < n) p[i] = rsqrtf(p[i]);               // deg >= 1 always
}

// ---------------------------------------------------------------------------
// out[n,f] = h[n,f]*dinv[n]^2 + b[f]  (self loop + bias; full overwrite, f4)
// ---------------------------------------------------------------------------
__global__ void k_node_init(const float4* __restrict__ H4, const float* __restrict__ dinv,
                            const float4* __restrict__ b4, float4* __restrict__ out4,
                            int total4) {
  int i = blockIdx.x * blockDim.x + threadIdx.x;
  if (i >= total4) return;
  int node = i >> 5;                            // 32 float4 per node
  int f4   = i & 31;
  float di = dinv[node];
  float s  = di * di;
  float4 h = H4[i], bb = b4[f4];
  float4 o;
  o.x = fmaf(h.x, s, bb.x); o.y = fmaf(h.y, s, bb.y);
  o.z = fmaf(h.z, s, bb.z); o.w = fmaf(h.w, s, bb.w);
  out4[i] = o;
}

// ---------------------------------------------------------------------------
// Edge scatter: out[dst] += h[src] * dinv[src]*dinv[dst]
// One wave per edge, float4 per lane; grid-stride with lookahead prefetch.
// ---------------------------------------------------------------------------
__global__ __launch_bounds__(256) void k_edge_scatter(
    const int* __restrict__ ei, const float* __restrict__ H,
    const float* __restrict__ dinv, float* __restrict__ out, int E)
{
  const int lane = threadIdx.x & 31;
  const int waves_total = gridDim.x * (blockDim.x >> 5);
  int e = blockIdx.x * (blockDim.x >> 5) + (threadIdx.x >> 5);

  for (; e < E; e += waves_total) {
    int s = ei[e];
    int d = ei[E + e];

    int e2 = e + waves_total;                   // hide gather latency
    if (e2 < E) {
      int s2 = ei[e2];
      __builtin_prefetch(H + (size_t)s2 * FEAT + lane * 4, 0, 0);
    }

    float nrm = dinv[s] * dinv[d];
    const float4 v = ((const float4*)(H + (size_t)s * FEAT))[lane];
    float* od = out + (size_t)d * FEAT + lane * 4;
    atomicAdd(od + 0, v.x * nrm);
    atomicAdd(od + 1, v.y * nrm);
    atomicAdd(od + 2, v.z * nrm);
    atomicAdd(od + 3, v.w * nrm);
  }
}

// ---------------------------------------------------------------------------
// Host orchestration
// ---------------------------------------------------------------------------
extern "C" void kernel_launch(void* const* d_in, const int* in_sizes, int n_in,
                              void* d_out, int out_size, void* d_ws, size_t ws_size,
                              hipStream_t stream) {
  const float* x  = (const float*)d_in[0];
  const int*   ei = (const int*)  d_in[1];
  const float* W1 = (const float*)d_in[2];
  const float* b1 = (const float*)d_in[3];
  const float* W2 = (const float*)d_in[4];
  const float* b2 = (const float*)d_in[5];
  const float* W3 = (const float*)d_in[6];
  const float* b3 = (const float*)d_in[7];

  const int N = in_sizes[0] / FEAT;
  const int E = in_sizes[1] / 2;
  const int total  = N * FEAT;
  const int total4 = total / 4;

  // Workspace layout (256B aligned): dinv[N] | h[N*128] | bufA[N*128]
  char* ws = (char*)d_ws;
  size_t off = 0;
  float* dinv = (float*)(ws + off); off += ((size_t)N * 4 + 255) & ~(size_t)255;
  float* h    = (float*)(ws + off); off += ((size_t)total * 4 + 255) & ~(size_t)255;
  float* bufA = (float*)(ws + off);
  float* out  = (float*)d_out;

  const int T = 256;
  const dim3 gN((N + T - 1) / T), gE((E + T - 1) / T), gF4((total4 + T - 1) / T);
  const dim3 gGemm((N + 127) / 128);
  const dim3 gScat(4096);

  // Degrees -> dinv
  k_fill_ones<<<gN, T, 0, stream>>>(dinv, N);
  k_deg_acc <<<gE, T, 0, stream>>>(ei, dinv, E);
  k_rsqrt   <<<gN, T, 0, stream>>>(dinv, N);

  // Layer 1: h = x@W1 ; bufA = self+bias ; scatter edges  (ReLU folded into L2 gemm)
  gemm_wmma_bf16<0><<<gGemm, T, 0, stream>>>(x, W1, h, N);
  k_node_init   <<<gF4, T, 0, stream>>>((const float4*)h, dinv, (const float4*)b1,
                                        (float4*)bufA, total4);
  k_edge_scatter<<<gScat, T, 0, stream>>>(ei, h, dinv, bufA, E);

  // Layer 2 (relu on input): h = relu(bufA)@W2 ; bufA = self+bias ; scatter
  gemm_wmma_bf16<1><<<gGemm, T, 0, stream>>>(bufA, W2, h, N);
  k_node_init   <<<gF4, T, 0, stream>>>((const float4*)h, dinv, (const float4*)b2,
                                        (float4*)bufA, total4);
  k_edge_scatter<<<gScat, T, 0, stream>>>(ei, h, dinv, bufA, E);

  // Layer 3 (relu on input, no output relu): write straight into d_out
  gemm_wmma_bf16<1><<<gGemm, T, 0, stream>>>(bufA, W3, h, N);
  k_node_init   <<<gF4, T, 0, stream>>>((const float4*)h, dinv, (const float4*)b3,
                                        (float4*)out, total4);
  k_edge_scatter<<<gScat, T, 0, stream>>>(ei, h, dinv, out, E);
}
